// SelfAttentionHead_833223655754
// MI455X (gfx1250) — compile-verified, matrix operand order checked by reading, not run on hardware
//
#include <hip/hip_runtime.h>
#include <hip/hip_bf16.h>

// ---- problem constants (match reference) ----
static constexpr int B = 4;
static constexpr int T = 2048;
static constexpr int C = 1024;
static constexpr int H = 64;
static constexpr float SM_SCALE = 0.03125f; // C^-0.5 = 1/32 (reference scales by C, not H)

typedef __attribute__((ext_vector_type(16))) _Float16 v16h;
typedef __attribute__((ext_vector_type(8)))  _Float16 v8h;
typedef __attribute__((ext_vector_type(8)))  float    v8f;

// ---------------------------------------------------------------------------
// WMMA fragment helpers for V_WMMA_F32_16X16X32_F16 (wave32).
// A (16x32 f16): lane<16 -> row=lane,    K = {ko..ko+7} U {16+ko..16+ko+7}, ko=0
//                lane>=16 -> row=lane-16, same with ko=8
// B (32x16 f16): mirrored with N in place of M, so loading "rows" of a
//                row-major [N x K] matrix serves B^T-style operands.
// C/D (16x16 f32, 8 VGPRs): element r of lane l = C[(l>>4)*8 + r][l & 15]
// ---------------------------------------------------------------------------

__device__ __forceinline__ v16h frag_from_f32_rows(const float* __restrict__ base,
                                                   int ld, int r0, int k0, int lane) {
  const int row = lane & 15;
  const int ko  = (lane >> 4) * 8;
  const float* p = base + (size_t)(r0 + row) * ld + k0 + ko;
  v16h a;
#pragma unroll
  for (int i = 0; i < 8; ++i) {
    a[i]     = (_Float16)p[i];
    a[8 + i] = (_Float16)p[16 + i];
  }
  return a;
}

__device__ __forceinline__ v16h frag_from_f16_rows(const _Float16* __restrict__ base,
                                                   int ld, int r0, int k0, int lane) {
  const int row = lane & 15;
  const int ko  = (lane >> 4) * 8;
  const _Float16* p = base + (size_t)(r0 + row) * ld + k0 + ko;
  v8h lo = *(const v8h*)(p);       // 16B aligned: ld mult of 8 halves, k0+ko mult of 8
  v8h hi = *(const v8h*)(p + 16);
  v16h a;
#pragma unroll
  for (int i = 0; i < 8; ++i) { a[i] = lo[i]; a[8 + i] = hi[i]; }
  return a;
}

__device__ __forceinline__ v8f wmma_f16(v16h a, v16h b, v8f c) {
  return __builtin_amdgcn_wmma_f32_16x16x32_f16(false, a, false, b,
                                                (short)0, c, false, false);
}

__device__ __forceinline__ float rowmax16(float v) {
#pragma unroll
  for (int m = 1; m <= 8; m <<= 1) v = fmaxf(v, __shfl_xor(v, m, 32));
  return v;
}
__device__ __forceinline__ float rowsum16(float v) {
#pragma unroll
  for (int m = 1; m <= 8; m <<= 1) v += __shfl_xor(v, m, 32);
  return v;
}

// ---------------------------------------------------------------------------
// Kernel 0: one-shot fp32 -> f16 conversion of [Wq;Wk;Wv] into a [192 x C] blob
// ---------------------------------------------------------------------------
__global__ __launch_bounds__(256) void wconv_kernel(
    const float* __restrict__ Wq, const float* __restrict__ Wk,
    const float* __restrict__ Wv, _Float16* __restrict__ wcat) {
  const int idx = blockIdx.x * 256 + threadIdx.x; // 0 .. 3*H*C-1
  const int mat = idx / (H * C);
  const int off = idx - mat * (H * C);
  const float* src = (mat == 0) ? Wq : (mat == 1) ? Wk : Wv;
  wcat[idx] = (_Float16)src[off];
}

// ---------------------------------------------------------------------------
// Kernel 1: fused Q/K/V projection: one sweep over x feeds all three GEMMs.
// M = B*T, K = C, N = 3*H (= rows of wcat). Wave -> 16-row x-tile, 12 accs.
// grid = (M/(16*8)), block = 256 (8 waves).
// ---------------------------------------------------------------------------
__global__ __launch_bounds__(256) void qkv_proj_kernel(
    const float* __restrict__ x,
    const _Float16* __restrict__ wcat,  // [3*H, C] f16
    _Float16* __restrict__ qb,
    _Float16* __restrict__ kb,
    _Float16* __restrict__ vb) {
  const int lane = threadIdx.x & 31;
  const int wave = threadIdx.x >> 5;
  const int m0   = (blockIdx.x * 8 + wave) * 16;

  v8f acc[3][4];
#pragma unroll
  for (int p = 0; p < 3; ++p)
#pragma unroll
    for (int j = 0; j < 4; ++j) acc[p][j] = 0;

  for (int k0 = 0; k0 < C; k0 += 32) {
    v16h a = frag_from_f32_rows(x, C, m0, k0, lane); // converted once, used 12x
#pragma unroll
    for (int p = 0; p < 3; ++p)
#pragma unroll
      for (int j = 0; j < 4; ++j) {
        v16h bfr = frag_from_f16_rows(wcat, C, p * 64 + j * 16, k0, lane);
        acc[p][j] = wmma_f16(a, bfr, acc[p][j]);
      }
  }

  const int colL  = lane & 15;
  const int rbase = (lane >> 4) * 8;
  _Float16* outs[3] = {qb, kb, vb};
#pragma unroll
  for (int p = 0; p < 3; ++p)
#pragma unroll
    for (int j = 0; j < 4; ++j)
#pragma unroll
      for (int r = 0; r < 8; ++r)
        outs[p][(size_t)(m0 + rbase + r) * H + j * 16 + colL] =
            (_Float16)acc[p][j][r];
}

// ---------------------------------------------------------------------------
// Kernel 2: causal flash attention forward.
// One wave per 16-row query tile; 4 waves/block. V tiles staged to LDS with
// CDNA5 async global->LDS copies (ASYNCcnt), overlapping QK^T + softmax.
// grid = (T/64, B), block = 128.
// ---------------------------------------------------------------------------
#define ASYNC_CP_B128(OFF)                                                  \
  asm volatile("global_load_async_to_lds_b128 %0, %1, off offset:" #OFF    \
               :: "v"(ldsVAddr), "v"(gAddr) : "memory")

__global__ __launch_bounds__(128) void attn_fwd_kernel(
    const _Float16* __restrict__ qbuf,
    const _Float16* __restrict__ kbuf,
    const _Float16* __restrict__ vbuf,
    float* __restrict__ out) {
  __shared__ __align__(16) _Float16 ldsP[4][16][32]; // P staging (C- -> A-layout)
  __shared__ __align__(16) _Float16 ldsV[4][32][64]; // async-staged V tiles

  const int lane = threadIdx.x & 31;
  const int wave = threadIdx.x >> 5;
  const int b    = blockIdx.y;
  const int q0   = (blockIdx.x * 4 + wave) * 16;

  const _Float16* qp = qbuf + (size_t)b * T * H;
  const _Float16* kp = kbuf + (size_t)b * T * H;
  const _Float16* vp = vbuf + (size_t)b * T * H;

  const int colL  = lane & 15;
  const int rbase = (lane >> 4) * 8;
  const int ko    = rbase; // same (lane>>4)*8 offset used by A-layout loads

  // per-lane LDS dest: lane L owns V row L of this wave's tile (128 B/row)
  const unsigned ldsVAddr =
      (unsigned)(uintptr_t)(&ldsV[wave][0][0]) + (unsigned)(lane * (H * 2));

  // Q tile A-fragments (K-dim = H = 64 -> two k-steps)
  const v16h aQ0 = frag_from_f16_rows(qp, H, q0, 0,  lane);
  const v16h aQ1 = frag_from_f16_rows(qp, H, q0, 32, lane);

  v8f accO[4];
#pragma unroll
  for (int j = 0; j < 4; ++j) accO[j] = 0;
  float mrow[8], lrow[8];
#pragma unroll
  for (int r = 0; r < 8; ++r) { mrow[r] = -1e30f; lrow[r] = 0.0f; }

  const int nTiles = (q0 + 47) >> 5; // keys [0, q0+16) covered by 32-wide tiles

  for (int t = 0; t < nTiles; ++t) {
    const int s0 = t * 32;

    // kick off async V-tile copy (32 rows x 128 B); lane L copies row s0+L
    {
      const unsigned long long gAddr =
          (unsigned long long)(uintptr_t)vp +
          (unsigned long long)(s0 + lane) * (H * 2);
      ASYNC_CP_B128(0);   ASYNC_CP_B128(16);  ASYNC_CP_B128(32);
      ASYNC_CP_B128(48);  ASYNC_CP_B128(64);  ASYNC_CP_B128(80);
      ASYNC_CP_B128(96);  ASYNC_CP_B128(112);
    }
    if (t + 1 < nTiles)
      __builtin_prefetch(kp + (size_t)(s0 + 32) * H, 0, 3); // global_prefetch_b8

    // S = (Q @ K^T) for two 16-wide key sub-tiles
    v8f S[2];
#pragma unroll
    for (int sub = 0; sub < 2; ++sub) {
      v16h bK0 = frag_from_f16_rows(kp, H, s0 + sub * 16, 0,  lane);
      v16h bK1 = frag_from_f16_rows(kp, H, s0 + sub * 16, 32, lane);
      v8f c = 0;
      c = wmma_f16(aQ0, bK0, c);
      c = wmma_f16(aQ1, bK1, c);
      S[sub] = c;
    }

    // scale + causal mask + online softmax (row = q0+rbase+r, col = s0[+16]+colL)
#pragma unroll
    for (int r = 0; r < 8; ++r) {
      const int row = q0 + rbase + r;
      float s0v = S[0][r] * SM_SCALE;
      float s1v = S[1][r] * SM_SCALE;
      if (s0 + colL      > row) s0v = -1e30f;
      if (s0 + 16 + colL > row) s1v = -1e30f;

      float mnew = fmaxf(mrow[r], rowmax16(fmaxf(s0v, s1v)));
      float e0 = __expf(s0v - mnew);
      float e1 = __expf(s1v - mnew);
      float alpha = __expf(mrow[r] - mnew);
      lrow[r] = lrow[r] * alpha + rowsum16(e0 + e1);
      mrow[r] = mnew;
#pragma unroll
      for (int j = 0; j < 4; ++j) accO[j][r] *= alpha;

      ldsP[wave][rbase + r][colL]      = (_Float16)e0;
      ldsP[wave][rbase + r][16 + colL] = (_Float16)e1;
    }

    asm volatile("s_wait_dscnt 0" ::: "memory");     // P stores visible
    asm volatile("s_wait_asynccnt 0" ::: "memory");  // V tile landed in LDS

    // reload P in A-layout from LDS (16-byte ds loads)
    v16h aP;
    {
      const _Float16* pr = &ldsP[wave][colL][0];
      v8h lo = *(const v8h*)(pr + ko);
      v8h hi = *(const v8h*)(pr + 16 + ko);
#pragma unroll
      for (int i = 0; i < 8; ++i) { aP[i] = lo[i]; aP[8 + i] = hi[i]; }
    }

    // O += P @ V  (V B-fragments gathered from LDS-staged tile)
#pragma unroll
    for (int j = 0; j < 4; ++j) {
      const int h = j * 16 + colL;
      v16h bV;
#pragma unroll
      for (int i = 0; i < 8; ++i) {
        bV[i]     = ldsV[wave][ko + i][h];
        bV[8 + i] = ldsV[wave][16 + ko + i][h];
      }
      accO[j] = wmma_f16(aP, bV, accO[j]);
    }
    asm volatile("s_wait_dscnt 0" ::: "memory"); // LDS reads done before next iter
  }

  // normalize and write fp32 output [B, T, H]
#pragma unroll
  for (int r = 0; r < 8; ++r) {
    const float inv = 1.0f / lrow[r];
#pragma unroll
    for (int j = 0; j < 4; ++j)
      out[((size_t)b * T + q0 + rbase + r) * H + j * 16 + colL] = accO[j][r] * inv;
  }
}

// ---------------------------------------------------------------------------
extern "C" void kernel_launch(void* const* d_in, const int* in_sizes, int n_in,
                              void* d_out, int out_size, void* d_ws, size_t ws_size,
                              hipStream_t stream) {
  const float* x  = (const float*)d_in[0];
  const float* Wq = (const float*)d_in[1];
  const float* Wk = (const float*)d_in[2];
  const float* Wv = (const float*)d_in[3];
  float* out = (float*)d_out;

  _Float16* qb   = (_Float16*)d_ws;
  _Float16* kb   = qb + (size_t)B * T * H;
  _Float16* vb   = kb + (size_t)B * T * H;
  _Float16* wcat = vb + (size_t)B * T * H;  // [3*H, C] f16

  // 0) convert weights once (3*H*C = 196608 elements)
  wconv_kernel<<<(3 * H * C) / 256, 256, 0, stream>>>(Wq, Wk, Wv, wcat);

  // 1) fused Q/K/V projection: one sweep of x
  qkv_proj_kernel<<<(B * T) / (16 * 8), 256, 0, stream>>>(x, wcat, qb, kb, vb);

  // 2) flash attention: 4 query tiles (of 16 rows) per block
  dim3 gAttn(T / 64, B);
  attn_fwd_kernel<<<gAttn, 128, 0, stream>>>(qb, kb, vb, out);
}
